// BrainGNNBlock_76141180223558
// MI455X (gfx1250) — compile-verified, matrix operand order checked by reading, not run on hardware
//
#include <hip/hip_runtime.h>
#include <hip/hip_bf16.h>

#define NN 50000
#define EE 800000
#define DD 256
#define CC 16
#define KK 25000
#define BINS 4096
#define ITEMS 4
#define SCAN_TILE 1024
#define NB ((NN + SCAN_TILE - 1) / SCAN_TILE)
#define MT 64
#define TSTEPS 128   /* CC * (DD/32) K-steps of 32 */

typedef __attribute__((ext_vector_type(16))) __bf16 v16bf;
typedef __attribute__((ext_vector_type(8)))  float  v8f;

__device__ inline v16bf load_frag(const __bf16* p0, const __bf16* p1) {
    v16bf r;
    ((uint4*)&r)[0] = *(const uint4*)p0;
    ((uint4*)&r)[1] = *(const uint4*)p1;
    return r;
}

// ---------- basis weights f32 -> bf16 (2 MB, stays L2 resident) ----------
__global__ __launch_bounds__(256) void k_cvt_bw(const float* __restrict__ src,
                                                __bf16* __restrict__ dst) {
    int i = blockIdx.x * 256 + threadIdx.x;     // each thread converts 4
    float4 v = ((const float4*)src)[i];
    __bf16 t[4] __attribute__((aligned(8)));
    t[0] = (__bf16)v.x; t[1] = (__bf16)v.y; t[2] = (__bf16)v.z; t[3] = (__bf16)v.w;
    *(uint2*)&dst[i * 4] = *(const uint2*)t;
}

// ---------- WMMA GEMM: y[n,i] = sum_{c,j} re[n,c]*x[n,j]*B[c,i,j] ----------
// grid = (ceil(N/64), 2): blockIdx.y selects 128-column half of the output.
// B^T tiles stream into a 3-deep LDS ring via async global->LDS loads
// (ASYNCcnt); A fragments are built in registers straight from the LDS x-tile.
__global__ __launch_bounds__(256) void k_gemm(
    const float* __restrict__ x, const int* __restrict__ ri,
    const float* __restrict__ remb, const __bf16* __restrict__ bw,
    float* __restrict__ y)
{
    __shared__ __align__(16) __bf16 xs[MT][DD];        // 32 KB x tile (bf16)
    __shared__ __align__(16) float  res[MT][CC];       //  4 KB region-embed tile
    __shared__ __align__(16) __bf16 bts[3][128][32];   // 24 KB B^T ring buffer
    const int tid = threadIdx.x;
    const int m0  = blockIdx.x * MT;
    const int nb  = blockIdx.y;                        // column half (0/1)

    for (int i = tid; i < MT * (DD / 4); i += 256) {
        int m = i >> 6, q = i & 63;
        int gm = m0 + m;
        float4 v = make_float4(0.f, 0.f, 0.f, 0.f);
        if (gm < NN) v = ((const float4*)x)[gm * (DD / 4) + q];
        __bf16* d = &xs[m][q * 4];
        d[0] = (__bf16)v.x; d[1] = (__bf16)v.y; d[2] = (__bf16)v.z; d[3] = (__bf16)v.w;
    }
    for (int i = tid; i < MT * CC; i += 256) {
        int m = i >> 4, c = i & 15;
        int gm = m0 + m;
        res[m][c] = (gm < NN) ? remb[ri[gm] * CC + c] : 0.f;
    }

    // async-stage B^T tile for K-step `step` into ring slot `buf`:
    // row n (= output column within half) <- bw[c, nb*128+n, j0..j0+31]
    const int nr = tid >> 2;                // rows tid/4 and tid/4 + 64
    const int eo = (tid & 3) * 8;           // bf16 element offset in 64 B row
    auto issue_bt = [&](int step, int buf) {
        const int c = step >> 3, j0 = (step & 7) * 32;
#pragma unroll
        for (int r = 0; r < 2; ++r) {
            const int n = nr + r * 64;
            const __bf16* g = bw + ((size_t)(c * DD + nb * 128 + n)) * DD + j0 + eo;
            unsigned lds = (unsigned)(uintptr_t)(void*)&bts[buf][n][eo];
            asm volatile("global_load_async_to_lds_b128 %0, %1, off"
                         :: "v"(lds), "v"(g) : "memory");
        }
    };
    issue_bt(0, 0);
    issue_bt(1, 1);
    __syncthreads();                         // xs / res visible

    const int lane = tid & 31;
    const int wave = tid >> 5;
    const int mi   = wave & 3;               // 4 M-subtiles of 16 rows
    const int nh   = wave >> 2;              // 64-column half within this block
    const int lm   = lane & 15;
    const int koff = (lane < 16) ? 0 : 8;    // 16-bit A/B VGPR layout (ISA 7.12.2)
    const int row  = mi * 16 + lm;

    float rres[CC];
#pragma unroll
    for (int c = 0; c < CC; ++c) rres[c] = res[row][c];

    v8f acc[4];
#pragma unroll
    for (int t = 0; t < 4; ++t)
#pragma unroll
        for (int r = 0; r < 8; ++r) acc[t][r] = 0.f;

    for (int t = 0; t < TSTEPS; ++t) {
        // step-t group done (in-order completion); step-t+1 may stay in flight
        if (t + 1 < TSTEPS) asm volatile("s_wait_asynccnt 0x2" ::: "memory");
        else                asm volatile("s_wait_asynccnt 0x0" ::: "memory");
        __syncthreads();
        const int c  = t >> 3;
        const int j0 = (t & 7) * 32;
        const int buf = t % 3;
        const float s = rres[c];
        // A fragment in registers: a[k] = re[row,c] * x[row, j0+k]
        v16bf a;
        const __bf16* p0 = &xs[row][j0 + koff];
        const __bf16* p1 = &xs[row][j0 + koff + 16];
#pragma unroll
        for (int u = 0; u < 8; ++u) a[u]     = (__bf16)(s * (float)p0[u]);
#pragma unroll
        for (int u = 0; u < 8; ++u) a[8 + u] = (__bf16)(s * (float)p1[u]);
#pragma unroll
        for (int nt = 0; nt < 4; ++nt) {
            const int n = nh * 64 + nt * 16 + lm;
            v16bf b = load_frag(&bts[buf][n][koff], &bts[buf][n][koff + 16]);
            acc[nt] = __builtin_amdgcn_wmma_f32_16x16x32_bf16(
                false, a, false, b, (short)0, acc[nt], false, false);
        }
        // safe: every wave passed barrier(t) => all finished reading slot (t+2)%3
        if (t + 2 < TSTEPS) issue_bt(t + 2, (t + 2) % 3);
    }

    // epilogue: C layout — VGPR r: lanes0-15 M=r, lanes16-31 M=r+8; N = lane&15
    const int rbase = m0 + mi * 16 + ((lane >> 4) << 3);
#pragma unroll
    for (int nt = 0; nt < 4; ++nt) {
        const int col = nb * 128 + nh * 64 + nt * 16 + lm;
#pragma unroll
        for (int r = 0; r < 8; ++r) {
            const int rw = rbase + r;
            if (rw < NN) y[(size_t)rw * DD + col] = acc[nt][r];
        }
    }
}

// ---------- zero / init ----------
__global__ void k_zero_f4(float4* p, int n4) {
    for (int i = blockIdx.x * blockDim.x + threadIdx.x; i < n4; i += gridDim.x * blockDim.x)
        p[i] = make_float4(0.f, 0.f, 0.f, 0.f);
}
__global__ __launch_bounds__(256) void k_init_sel(int* hist, int* mask, unsigned* mm) {
    int i = blockIdx.x * 256 + threadIdx.x;
    if (i < BINS) hist[i] = 0;
    if (i < NN) mask[i] = 0;
    if (i == 0) { mm[0] = 0xFFFFFFFFu; mm[1] = 0u; }
}

// ---------- edge gate + scatter-add (memory-bound hot loop) ----------
__global__ __launch_bounds__(256) void k_edges(const int* __restrict__ eidx,
    const float* __restrict__ eattr, const float* __restrict__ y,
    float* __restrict__ outb, const float* __restrict__ ew, const float* __restrict__ eb)
{
    int e = blockIdx.x * 8 + (threadIdx.x >> 5);
    if (e >= EE) return;
    int lane = threadIdx.x & 31;
    int s = eidx[e], d = eidx[EE + e];
    float t = eattr[e] * ew[0] + eb[0];
    float g = 1.f / (1.f + __expf(-t));
    const float4* yr = (const float4*)(y + (size_t)s * DD);
    float* orow = outb + (size_t)d * DD;
#pragma unroll
    for (int h = 0; h < 2; ++h) {
        int q = lane + h * 32;
        float4 v = yr[q];
        atomicAdd(orow + q * 4 + 0, v.x * g);
        atomicAdd(orow + q * 4 + 1, v.y * g);
        atomicAdd(orow + q * 4 + 2, v.z * g);
        atomicAdd(orow + q * 4 + 3, v.w * g);
    }
}

// ---------- relu + per-column batch-norm stats (deterministic) ----------
__global__ __launch_bounds__(256) void k_stats(float* __restrict__ outb,
    float* __restrict__ meanv, float* __restrict__ invv)
{
    __shared__ float ss[256], sq[256];
    int d = blockIdx.x, tid = threadIdx.x;
    float s = 0.f, q = 0.f;
    for (int r = tid; r < NN; r += 256) {
        float v = outb[(size_t)r * DD + d];
        v = fmaxf(v, 0.f);
        outb[(size_t)r * DD + d] = v;
        s += v; q += v * v;
    }
    ss[tid] = s; sq[tid] = q; __syncthreads();
    for (int o = 128; o; o >>= 1) {
        if (tid < o) { ss[tid] += ss[tid + o]; sq[tid] += sq[tid + o]; }
        __syncthreads();
    }
    if (!tid) {
        float m = ss[0] / NN;
        float var = sq[0] / NN - m * m;
        meanv[d] = m;
        invv[d] = rsqrtf(var + 1e-5f);
    }
}

// ---------- normalize in place + node scores (one wave per node) ----------
__global__ __launch_bounds__(256) void k_norm_score(float* __restrict__ outb,
    const float* __restrict__ meanv, const float* __restrict__ invv,
    const float* __restrict__ gamma, const float* __restrict__ beta,
    const float* __restrict__ sw, const float* __restrict__ sb,
    float* __restrict__ scores)
{
    int n = blockIdx.x * 8 + (threadIdx.x >> 5);
    int lane = threadIdx.x & 31;
    if (n >= NN) return;
    float4* row = (float4*)(outb + (size_t)n * DD);
    float dot = 0.f;
#pragma unroll
    for (int h = 0; h < 2; ++h) {
        int q = lane + h * 32;
        float4 v = row[q];
        float4 mn = ((const float4*)meanv)[q];
        float4 iv = ((const float4*)invv)[q];
        float4 g  = ((const float4*)gamma)[q];
        float4 be = ((const float4*)beta)[q];
        float4 w4 = ((const float4*)sw)[q];
        v.x = (v.x - mn.x) * iv.x * g.x + be.x; dot += v.x * w4.x;
        v.y = (v.y - mn.y) * iv.y * g.y + be.y; dot += v.y * w4.y;
        v.z = (v.z - mn.z) * iv.z * g.z + be.z; dot += v.z * w4.z;
        v.w = (v.w - mn.w) * iv.w * g.w + be.w; dot += v.w * w4.w;
        row[q] = v;
    }
    for (int m2 = 16; m2; m2 >>= 1) dot += __shfl_xor(dot, m2);
    if (!lane) scores[n] = dot + sb[0];
}

// ---------- top-k via histogram threshold + deterministic scan placement ----------
__device__ inline unsigned fkey(float f) {
    unsigned b = __float_as_uint(f);
    return (b & 0x80000000u) ? ~b : (b | 0x80000000u);
}
__device__ inline float fkey_inv(unsigned k) {
    unsigned b = (k & 0x80000000u) ? (k & 0x7FFFFFFFu) : ~k;
    return __uint_as_float(b);
}
__device__ inline int sbin(float s, float smin, float sinv) {
    int b = (int)((s - smin) * sinv);
    return b < 0 ? 0 : (b > BINS - 1 ? BINS - 1 : b);
}

__global__ __launch_bounds__(256) void k_minmax(const float* __restrict__ scores, unsigned* mm) {
    __shared__ unsigned smn[256], smx[256];
    int tid = threadIdx.x;
    unsigned mn = 0xFFFFFFFFu, mx = 0u;
    for (int i = blockIdx.x * 256 + tid; i < NN; i += gridDim.x * 256) {
        unsigned k = fkey(scores[i]);
        mn = k < mn ? k : mn;
        mx = k > mx ? k : mx;
    }
    smn[tid] = mn; smx[tid] = mx; __syncthreads();
    for (int o = 128; o; o >>= 1) {
        if (tid < o) {
            smn[tid] = smn[tid + o] < smn[tid] ? smn[tid + o] : smn[tid];
            smx[tid] = smx[tid + o] > smx[tid] ? smx[tid + o] : smx[tid];
        }
        __syncthreads();
    }
    if (!tid) { atomicMin(&mm[0], smn[0]); atomicMax(&mm[1], smx[0]); }
}

__global__ __launch_bounds__(256) void k_hist(const float* scores, const unsigned* mm, int* hist) {
    float smin = fkey_inv(mm[0]), smax = fkey_inv(mm[1]);
    float sinv = (float)BINS / fmaxf(smax - smin, 1e-30f);
    int i = blockIdx.x * 256 + threadIdx.x;
    if (i < NN) atomicAdd(&hist[sbin(scores[i], smin, sinv)], 1);
}

__global__ void k_scan_hist(const int* hist, int* meta) {
    if (threadIdx.x == 0 && blockIdx.x == 0) {
        int acc = 0, T = 0;
        for (int b = BINS - 1; b >= 0; --b) {
            if (acc + hist[b] >= KK) { T = b; break; }
            acc += hist[b];
        }
        meta[0] = T; meta[1] = acc; meta[2] = KK - acc;
    }
}

__global__ __launch_bounds__(256) void k_count(const float* scores, const unsigned* mm,
                                               const int* meta, int* bsum) {
    __shared__ int sc[256];
    float smin = fkey_inv(mm[0]), smax = fkey_inv(mm[1]);
    float sinv = (float)BINS / fmaxf(smax - smin, 1e-30f);
    int T = meta[0];
    int tid = threadIdx.x;
    int base = blockIdx.x * SCAN_TILE + tid * ITEMS;
    int hi = 0, eq = 0;
    for (int it = 0; it < ITEMS; ++it) {
        int n = base + it;
        if (n < NN) { int b = sbin(scores[n], smin, sinv); hi += (b > T); eq += (b == T); }
    }
    sc[tid] = (hi << 16) | eq; __syncthreads();
    for (int o = 128; o; o >>= 1) { if (tid < o) sc[tid] += sc[tid + o]; __syncthreads(); }
    if (!tid) { bsum[blockIdx.x * 2] = sc[0] >> 16; bsum[blockIdx.x * 2 + 1] = sc[0] & 0xFFFF; }
}

__global__ void k_scan_blocks(const int* bsum, int* boff) {
    if (threadIdx.x == 0 && blockIdx.x == 0) {
        int ah = 0, ae = 0;
        for (int i = 0; i < NB; ++i) {
            boff[i * 2] = ah; boff[i * 2 + 1] = ae;
            ah += bsum[i * 2]; ae += bsum[i * 2 + 1];
        }
    }
}

__global__ __launch_bounds__(256) void k_place(const float* scores, const unsigned* mm,
    const int* meta, const int* boff, int* topk, int* mask)
{
    __shared__ int sc[256];
    float smin = fkey_inv(mm[0]), smax = fkey_inv(mm[1]);
    float sinv = (float)BINS / fmaxf(smax - smin, 1e-30f);
    int T = meta[0], above = meta[1], need = meta[2];
    int tid = threadIdx.x;
    int base = blockIdx.x * SCAN_TILE + tid * ITEMS;
    int hi = 0, eq = 0;
    int bins[ITEMS];
    for (int it = 0; it < ITEMS; ++it) {
        int n = base + it;
        bins[it] = -1;
        if (n < NN) { bins[it] = sbin(scores[n], smin, sinv); hi += (bins[it] > T); eq += (bins[it] == T); }
    }
    sc[tid] = (hi << 16) | eq; __syncthreads();
    for (int o = 1; o < 256; o <<= 1) {
        int v = (tid >= o) ? sc[tid - o] : 0;
        __syncthreads();
        sc[tid] += v;
        __syncthreads();
    }
    int incl = sc[tid];
    int ehi = (incl >> 16) - hi + boff[blockIdx.x * 2];
    int eeq = (incl & 0xFFFF) - eq + boff[blockIdx.x * 2 + 1];
    for (int it = 0; it < ITEMS; ++it) {
        int n = base + it;
        if (n >= NN) break;
        if (bins[it] > T) { topk[ehi] = n; mask[n] = 1; ehi++; }
        else if (bins[it] == T) { if (eeq < need) { topk[above + eeq] = n; mask[n] = 1; } eeq++; }
    }
}

// ---------- pooled gathers + pass-through outputs ----------
__global__ __launch_bounds__(256) void k_gather(const int* topk, const float* outb,
    const int* ri, const int* batch, float* xpool, int* regp, int* batp)
{
    int p = blockIdx.x * 8 + (threadIdx.x >> 5);
    if (p >= KK) return;
    int lane = threadIdx.x & 31;
    int idx = topk[p];
    const float4* s = (const float4*)(outb + (size_t)idx * DD);
    float4* d = (float4*)(xpool + (size_t)p * DD);
    d[lane] = s[lane];
    d[lane + 32] = s[lane + 32];
    if (!lane) { regp[p] = ri[idx]; batp[p] = batch[idx]; }
}

__global__ __launch_bounds__(256) void k_edgeout(const int* eidx, const float* eattr,
    const int* mask, int* eio, float* eao, float* emo)
{
    int e = blockIdx.x * 256 + threadIdx.x;
    if (e >= EE) return;
    int s = eidx[e], d = eidx[EE + e];
    eio[e] = s; eio[EE + e] = d;
    eao[e] = eattr[e];
    emo[e] = (mask[s] && mask[d]) ? 1.f : 0.f;
}

extern "C" void kernel_launch(void* const* d_in, const int* in_sizes, int n_in,
                              void* d_out, int out_size, void* d_ws, size_t ws_size,
                              hipStream_t stream)
{
    const float* x     = (const float*)d_in[0];
    const int*   eidx  = (const int*)d_in[1];
    const float* eattr = (const float*)d_in[2];
    const int*   ri    = (const int*)d_in[3];
    const int*   batch = (const int*)d_in[4];
    const float* remb  = (const float*)d_in[5];
    const float* bwf   = (const float*)d_in[6];
    const float* ew    = (const float*)d_in[7];
    const float* eb    = (const float*)d_in[8];
    const float* gamma = (const float*)d_in[9];
    const float* beta  = (const float*)d_in[10];
    const float* sw    = (const float*)d_in[11];
    const float* sb    = (const float*)d_in[12];

    char* ws = (char*)d_ws;
    size_t off = 0;
    auto alloc = [&](size_t bytes) -> char* {
        char* p = ws + off;
        off += (bytes + 255) & ~(size_t)255;
        return p;
    };
    float*    y     = (float*)alloc((size_t)NN * DD * 4);
    float*    outb  = (float*)alloc((size_t)NN * DD * 4);
    __bf16*   bwb   = (__bf16*)alloc((size_t)CC * DD * DD * 2);
    float*    meanv = (float*)alloc(DD * 4);
    float*    invv  = (float*)alloc(DD * 4);
    unsigned* mm    = (unsigned*)alloc(2 * 4);
    int*      hist  = (int*)alloc(BINS * 4);
    int*      meta  = (int*)alloc(4 * 4);
    int*      bsum  = (int*)alloc(NB * 2 * 4);
    int*      boff  = (int*)alloc(NB * 2 * 4);
    int*      topk  = (int*)alloc(KK * 4);
    int*      mask  = (int*)alloc(NN * 4);

    // output packing (reference tuple order), 4-byte slots
    float* xpool  = (float*)d_out;
    int*   eio    = (int*)(xpool + (size_t)KK * DD);
    float* eao    = (float*)(eio + 2 * (size_t)EE);
    float* emo    = eao + EE;
    int*   regp   = (int*)(emo + EE);
    int*   batp   = regp + KK;
    float* scores = (float*)(batp + KK);

    k_cvt_bw<<<(CC * DD * DD) / 4 / 256, 256, 0, stream>>>(bwf, bwb);
    k_zero_f4<<<2048, 256, 0, stream>>>((float4*)outb, NN * DD / 4);
    k_init_sel<<<(NN + 255) / 256, 256, 0, stream>>>(hist, mask, mm);
    k_gemm<<<dim3((NN + MT - 1) / MT, 2), 256, 0, stream>>>(x, ri, remb, bwb, y);
    k_edges<<<(EE + 7) / 8, 256, 0, stream>>>(eidx, eattr, y, outb, ew, eb);
    k_stats<<<DD, 256, 0, stream>>>(outb, meanv, invv);
    k_norm_score<<<(NN + 7) / 8, 256, 0, stream>>>(outb, meanv, invv, gamma, beta, sw, sb, scores);
    k_minmax<<<256, 256, 0, stream>>>(scores, mm);
    k_hist<<<(NN + 255) / 256, 256, 0, stream>>>(scores, mm, hist);
    k_scan_hist<<<1, 1, 0, stream>>>(hist, meta);
    k_count<<<NB, 256, 0, stream>>>(scores, mm, meta, bsum);
    k_scan_blocks<<<1, 1, 0, stream>>>(bsum, boff);
    k_place<<<NB, 256, 0, stream>>>(scores, mm, meta, boff, topk, mask);
    k_gather<<<(KK + 7) / 8, 256, 0, stream>>>(topk, outb, ri, batch, xpool, regp, batp);
    k_edgeout<<<(EE + 255) / 256, 256, 0, stream>>>(eidx, eattr, mask, eio, eao, emo);
}
